// VideoVisionTransformer_38757784879860
// MI455X (gfx1250) — matrix operand order repaired and checked by reading, not compiled, not run on hardware
//
#include <hip/hip_runtime.h>
#include <hip/hip_bf16.h>
#include <math.h>

// ---------------- model constants ----------------
constexpr int Bc    = 16;
constexpr int Tc    = 16;
constexpr int Cc    = 3;
constexpr int Hc    = 112;
constexpr int Wc    = 112;
constexpr int Pc    = 16;
constexpr int Dc    = 768;
constexpr int NHc   = 12;
constexpr int HDc   = 64;
constexpr int DEPTHc= 12;
constexpr int MLPHc = 3072;
constexpr int NCc   = 400;
constexpr int Kc    = 4;          // ceil(sqrt(T))
constexpr int HPc   = Kc*Hc/Pc;   // 28
constexpr int WPc   = Kc*Wc/Pc;   // 28
constexpr int Nc    = HPc*WPc;    // 784 patches
constexpr int Sc    = Nc + 1;     // 785 tokens
constexpr int SPc   = 800;        // padded tokens (mult of 32)
constexpr int SCPc  = 808;        // LDS row stride for scores
constexpr int Mrows = Bc*SPc;     // 12800 padded activation rows

typedef __attribute__((ext_vector_type(16))) __bf16 bf16x16;
typedef __attribute__((ext_vector_type(8)))  __bf16 bf16x8;
typedef __attribute__((ext_vector_type(8)))  float  f32x8;
typedef __attribute__((ext_vector_type(4)))  unsigned int u32x4;
typedef __attribute__((ext_vector_type(8)))  int    i32x8;
typedef __attribute__((ext_vector_type(4)))  int    i32x4;

__device__ __forceinline__ __bf16 f2bf(float f) {
  unsigned u = __builtin_bit_cast(unsigned, f);
  unsigned r = u + 0x7fffu + ((u >> 16) & 1u);   // round-to-nearest-even
  unsigned short hs = (unsigned short)(r >> 16);
  return __builtin_bit_cast(__bf16, hs);
}

__device__ __forceinline__ void stoval(float*  p, float v) { *p = v; }
__device__ __forceinline__ void stoval(__bf16* p, float v) { *p = f2bf(v); }

// =====================================================================
// Weight pack: W (Kdim x Ndim, row-major f32)  ->  WMMA B-fragment order
// packed[((kt*NT + nt)*32 + lane)*16 + j] = bf16(W[kt*32 + (lane/16)*16+j][nt*16 + lane%16])
// =====================================================================
__global__ __launch_bounds__(256)
void vivit_pack_b(const float* __restrict__ W, __bf16* __restrict__ Bp,
                  int Kdim, int Ndim, int NT)
{
  long total = (long)(Kdim >> 5) * NT * 512;
  long idx = (long)blockIdx.x * 256 + threadIdx.x;
  if (idx >= total) return;
  int  j    = (int)(idx & 15);
  int  lane = (int)((idx >> 4) & 31);
  long tile = idx >> 9;
  int  nt   = (int)(tile % NT);
  int  kt   = (int)(tile / NT);
  int  kd   = (lane >> 4) * 16 + j;
  int  col  = nt * 16 + (lane & 15);
  int  krow = kt * 32 + kd;
  float v = (col < Ndim) ? W[(long)krow * Ndim + col] : 0.f;
  Bp[idx] = f2bf(v);
}

// =====================================================================
// Generic WMMA GEMM:  out[M,N] = A[M,K](bf16) @ Bpacked + bias (+res) (+gelu)
// Block 256 = 8 waves (2 x 4); wave tile 32x64; block tile 64x256.
// 8 WMMAs per (2 A-frag + 4 B-frag) loads -> 2x the math per byte vs 16x64.
// =====================================================================
enum { EPI_F32_BIAS = 0, EPI_F32_BIAS_RES = 1, EPI_BF16_BIAS = 2, EPI_BF16_BIAS_GELU = 3 };

template<int EPI>
__global__ __launch_bounds__(256)
void vivit_gemm_wmma(const __bf16* __restrict__ A, const __bf16* __restrict__ Bp,
                     const float* __restrict__ bias, const float* __restrict__ res,
                     void* __restrict__ outp, int M, int N, int Kd, int NT)
{
  const int lane = threadIdx.x & 31;
  const int wv   = threadIdx.x >> 5;
  const int mw   = wv & 1;           // 2 waves along M
  const int nw   = wv >> 1;          // 4 waves along N
  const int tileM  = blockIdx.y * 64 + mw * 32;
  const int ntBase = blockIdx.x * 16 + nw * 4;
  const int half = lane >> 4;
  const int l16  = lane & 15;

  int ar0 = tileM + l16;      if (ar0 > M - 1) ar0 = M - 1;   // clamp small-M tails
  int ar1 = tileM + 16 + l16; if (ar1 > M - 1) ar1 = M - 1;
  const __bf16* ap0 = A + (size_t)ar0 * Kd + half * 16;
  const __bf16* ap1 = A + (size_t)ar1 * Kd + half * 16;
  const __bf16* bkl = Bp + (size_t)lane * 16 + (size_t)ntBase * 512;

  f32x8 acc[2][4] = {};
  const int KT = Kd >> 5;
  for (int kt = 0; kt < KT; ++kt) {
    bf16x8 a0lo = *(const bf16x8*)(ap0);
    bf16x8 a0hi = *(const bf16x8*)(ap0 + 8);
    bf16x8 a1lo = *(const bf16x8*)(ap1);
    bf16x8 a1hi = *(const bf16x8*)(ap1 + 8);
    ap0 += 32; ap1 += 32;
    __builtin_prefetch(ap0 + 96, 0, 0);
    __builtin_prefetch(ap1 + 96, 0, 0);
    bf16x16 af0, af1;
#pragma unroll
    for (int i = 0; i < 8; ++i) {
      af0[i] = a0lo[i]; af0[i + 8] = a0hi[i];
      af1[i] = a1lo[i]; af1[i + 8] = a1hi[i];
    }
    const __bf16* bt = bkl + (size_t)kt * NT * 512;
#pragma unroll
    for (int j = 0; j < 4; ++j) {
      bf16x16 bfv = *(const bf16x16*)(bt + (size_t)j * 512);
      acc[0][j] = __builtin_amdgcn_wmma_f32_16x16x32_bf16(
                    false, af0, false, bfv, (short)0, acc[0][j], false, false);
      acc[1][j] = __builtin_amdgcn_wmma_f32_16x16x32_bf16(
                    false, af1, false, bfv, (short)0, acc[1][j], false, false);
    }
  }

#pragma unroll
  for (int mi = 0; mi < 2; ++mi) {
#pragma unroll
    for (int j = 0; j < 4; ++j) {
      int col = (ntBase + j) * 16 + l16;
      if (col >= N) continue;
      float bc = bias[col];
#pragma unroll
      for (int r = 0; r < 8; ++r) {
        int row = tileM + mi * 16 + half * 8 + r;
        if (row >= M) continue;
        size_t o = (size_t)row * N + col;
        float v = acc[mi][j][r] + bc;
        if (EPI == EPI_F32_BIAS_RES) v += res[o];
        if (EPI == EPI_BF16_BIAS_GELU) v = 0.5f * v * (1.f + erff(v * 0.70710678118f));
        if (EPI == EPI_F32_BIAS || EPI == EPI_F32_BIAS_RES) ((float*)outp)[o] = v;
        else ((__bf16*)outp)[o] = f2bf(v);
      }
    }
  }
}

// =====================================================================
// TDM L2-warm: pull the next layer's packed weights through GL2 using the
// Tensor Data Mover (tensor_load_to_lds, 1-D tile descriptor, 32KB/block).
// LDS contents are throwaway; purpose is prefetch + exercising TENSORcnt.
// =====================================================================
__global__ __launch_bounds__(32)
void vivit_l2_warm(const void* __restrict__ src, unsigned sliceBytes,
                   unsigned long totalBytes)
{
#if defined(__has_builtin)
#if __has_builtin(__builtin_amdgcn_tensor_load_to_lds)
  extern __shared__ char lbuf[];
  (void)lbuf;
  unsigned long off = (unsigned long)blockIdx.x * sliceBytes;
  if (off >= totalBytes) return;
  unsigned long rem = totalBytes - off;
  unsigned elems = (unsigned)(((rem < sliceBytes) ? rem : (unsigned long)sliceBytes) >> 3); // 8B units
  unsigned tile0 = (elems > 0xffffu) ? 0xffffu : elems;
  unsigned long ga = (unsigned long)src + off;

  u32x4 g0;
  g0[0] = 1u;                                        // count=1, user descriptor
  g0[1] = 0u;                                        // lds_addr = 0 (scratch LDS)
  g0[2] = (unsigned)(ga & 0xffffffffu);              // global_addr[31:0]
  g0[3] = (unsigned)((ga >> 32) & 0x01ffffffu)       // global_addr[56:32]
          | 0x80000000u;                             // type = 2 ("image")
  i32x8 g1;
  g1[0] = (3 << 16);                                 // data_size = 8 bytes
  g1[1] = (int)((elems & 0xffffu) << 16);            // tensor_dim0[15:0]
  g1[2] = (int)(((elems >> 16) & 0xffffu)            // tensor_dim0[31:16]
          | (1u << 16));                             // tensor_dim1 = 1
  g1[3] = (int)(tile0 << 16);                        // tile_dim0
  g1[4] = 0;                                         // tile_dim1/2 = 0
  g1[5] = (int)elems;                                // tensor_dim0_stride[31:0]
  g1[6] = 0;
  g1[7] = 0;
  i32x4 z4 = {0, 0, 0, 0};
#if defined(__clang_major__) && (__clang_major__ >= 23)
  i32x8 z8 = {0, 0, 0, 0, 0, 0, 0, 0};
  __builtin_amdgcn_tensor_load_to_lds(g0, g1, z4, z4, z8, 0);
#else
  __builtin_amdgcn_tensor_load_to_lds(g0, g1, z4, z4, 0);
#endif
#if __has_builtin(__builtin_amdgcn_s_wait_tensorcnt)
  __builtin_amdgcn_s_wait_tensorcnt(0);
#endif
#endif
#endif
}

// =====================================================================
// LayerNorm over last dim (D=768), one block per row.
// =====================================================================
template<typename OT>
__global__ __launch_bounds__(256)
void vivit_layernorm(const float* __restrict__ x, const float* __restrict__ g,
                     const float* __restrict__ b, OT* __restrict__ y, int D)
{
  const int row = blockIdx.x;
  const float* xr = x + (size_t)row * D;
  __shared__ float sred[256];
  float s = 0.f;
  for (int d = threadIdx.x; d < D; d += 256) s += xr[d];
  sred[threadIdx.x] = s; __syncthreads();
  for (int st = 128; st > 0; st >>= 1) {
    if (threadIdx.x < st) sred[threadIdx.x] += sred[threadIdx.x + st];
    __syncthreads();
  }
  float mu = sred[0] / D; __syncthreads();
  float v = 0.f;
  for (int d = threadIdx.x; d < D; d += 256) { float t = xr[d] - mu; v += t * t; }
  sred[threadIdx.x] = v; __syncthreads();
  for (int st = 128; st > 0; st >>= 1) {
    if (threadIdx.x < st) sred[threadIdx.x] += sred[threadIdx.x + st];
    __syncthreads();
  }
  float rstd = rsqrtf(sred[0] / D + 1e-5f);
  for (int d = threadIdx.x; d < D; d += 256)
    stoval(y + (size_t)row * D + d, (xr[d] - mu) * rstd * g[d] + b[d]);
}

// =====================================================================
// Split QKV (f32, bias applied) into bf16 q/k [B,NH,SP,HD] and vT [B,NH,HD,SP]
// =====================================================================
__global__ __launch_bounds__(256)
void vivit_qkv_split(const float* __restrict__ qkv,
                     __bf16* __restrict__ q, __bf16* __restrict__ k,
                     __bf16* __restrict__ vT)
{
  size_t idx = (size_t)blockIdx.x * 256 + threadIdx.x;
  size_t total = (size_t)Bc * SPc * Dc;
  if (idx >= total) return;
  int dm = (int)(idx % Dc);
  size_t rem = idx / Dc;
  int s = (int)(rem % SPc);
  int b = (int)(rem / SPc);
  int h = dm >> 6, hd = dm & 63;
  const float* src = qkv + ((size_t)b * SPc + s) * (3 * Dc);
  bool valid = (s < Sc);
  float qv = valid ? src[dm]            : 0.f;
  float kv = valid ? src[Dc + dm]       : 0.f;
  float vv = valid ? src[2 * Dc + dm]   : 0.f;
  size_t bh = (size_t)b * NHc + h;
  q [(bh * SPc + s) * HDc + hd] = f2bf(qv);
  k [(bh * SPc + s) * HDc + hd] = f2bf(kv);
  vT[(bh * HDc + hd) * SPc + s] = f2bf(vv);
}

// =====================================================================
// Fused attention: one block = (b, head, 16-row q-tile); 4 waves.
// Phase1 scores via WMMA -> LDS; Phase2 softmax in LDS; Phase3 P@V via WMMA.
// q A-fragments are loop-invariant -> loaded once.
// =====================================================================
__global__ __launch_bounds__(128)
void vivit_attention(const __bf16* __restrict__ q, const __bf16* __restrict__ k,
                     const __bf16* __restrict__ vT, __bf16* __restrict__ o,
                     float scale)
{
  extern __shared__ char smem[];
  float*  sc    = (float*)smem;                    // [16][SCPc] f32 scores
  __bf16* pb    = (__bf16*)(sc + 16 * SCPc);       // [16][SCPc] bf16 probs
  float*  red   = (float*)(pb + 16 * SCPc);        // [16][8]
  float*  rstat = red + 16 * 8;                    // [16]

  const int qt = blockIdx.x, h = blockIdx.y, b = blockIdx.z;
  const int tid = threadIdx.x;
  const int wv = tid >> 5, lane = tid & 31, half = lane >> 4, l16 = lane & 15;

  const __bf16* qb = q  + ((size_t)b * NHc + h) * (size_t)SPc * HDc;
  const __bf16* kb = k  + ((size_t)b * NHc + h) * (size_t)SPc * HDc;
  const __bf16* vb = vT + ((size_t)b * NHc + h) * (size_t)HDc * SPc;

  // ---- Phase 1: scores = q @ k^T (tile 16 x SPc) ----
  const __bf16* qrow = qb + (size_t)(qt * 16 + l16) * HDc + half * 16;
  bf16x16 aq[2];
#pragma unroll
  for (int kk = 0; kk < 2; ++kk) {                 // HD=64 -> two k=32 fragments
    bf16x8 alo = *(const bf16x8*)(qrow + kk * 32);
    bf16x8 ahi = *(const bf16x8*)(qrow + kk * 32 + 8);
#pragma unroll
    for (int i = 0; i < 8; ++i) { aq[kk][i] = alo[i]; aq[kk][i + 8] = ahi[i]; }
  }
  for (int nt = wv; nt < SPc / 16; nt += 4) {
    f32x8 acc = {};
    const __bf16* krow = kb + (size_t)(nt * 16 + l16) * HDc + half * 16;
#pragma unroll
    for (int kk = 0; kk < 2; ++kk) {
      bf16x16 bfv = *(const bf16x16*)(krow + kk * 32);
      acc = __builtin_amdgcn_wmma_f32_16x16x32_bf16(
              false, aq[kk], false, bfv, (short)0, acc, false, false);
    }
#pragma unroll
    for (int r = 0; r < 8; ++r) {
      int m  = r + half * 8;
      int tn = nt * 16 + l16;
      float val = acc[r] * scale;
      if (tn >= Sc) val = -1e30f;                  // mask padded keys
      sc[m * SCPc + tn] = val;
    }
  }
  __syncthreads();

  // ---- Phase 2: row softmax (16 rows x 8 threads each) ----
  {
    int rid = tid >> 3, c0 = tid & 7;
    float m = -3.4e38f;
    for (int c = c0; c < SPc; c += 8) m = fmaxf(m, sc[rid * SCPc + c]);
    red[rid * 8 + c0] = m;
    __syncthreads();
    if (tid < 16) {
      float mm = red[tid * 8];
      for (int i = 1; i < 8; ++i) mm = fmaxf(mm, red[tid * 8 + i]);
      rstat[tid] = mm;
    }
    __syncthreads();
    float rm = rstat[rid];
    float s = 0.f;
    for (int c = c0; c < SPc; c += 8) {
      float e = __expf(sc[rid * SCPc + c] - rm);
      sc[rid * SCPc + c] = e;
      s += e;
    }
    red[rid * 8 + c0] = s;
    __syncthreads();
    if (tid < 16) {
      float ss = 0.f;
      for (int i = 0; i < 8; ++i) ss += red[tid * 8 + i];
      rstat[tid] = 1.f / fmaxf(ss, 1e-20f);
    }
    __syncthreads();
    float ri = rstat[rid];
    for (int c = c0; c < SPc; c += 8)
      pb[rid * SCPc + c] = f2bf(sc[rid * SCPc + c] * ri);
  }
  __syncthreads();

  // ---- Phase 3: O = P @ V ; wave wv handles hd columns [wv*16, wv*16+16) ----
  {
    f32x8 acc = {};
    const __bf16* vrow = vb + (size_t)(wv * 16 + l16) * SPc + half * 16;
    const __bf16* prow = pb + (size_t)l16 * SCPc + half * 16;
    for (int kt = 0; kt < SPc / 32; ++kt) {
      bf16x8 alo = *(const bf16x8*)(prow + kt * 32);
      bf16x8 ahi = *(const bf16x8*)(prow + kt * 32 + 8);
      bf16x16 af;
#pragma unroll
      for (int i = 0; i < 8; ++i) { af[i] = alo[i]; af[i + 8] = ahi[i]; }
      bf16x16 bfv = *(const bf16x16*)(vrow + kt * 32);
      acc = __builtin_amdgcn_wmma_f32_16x16x32_bf16(
              false, af, false, bfv, (short)0, acc, false, false);
    }
#pragma unroll
    for (int r = 0; r < 8; ++r) {
      int tok = qt * 16 + r + half * 8;
      int d   = h * HDc + wv * 16 + l16;
      o[((size_t)b * SPc + tok) * Dc + d] = f2bf(acc[r]);
    }
  }
}

// =====================================================================
// Patch im2col -> bf16 A matrix [B*N, C*P*P]
// =====================================================================
__global__ __launch_bounds__(256)
void vivit_patchify(const float* __restrict__ x, __bf16* __restrict__ A)
{
  size_t idx = (size_t)blockIdx.x * 256 + threadIdx.x;
  size_t total = (size_t)Bc * Nc * (Cc * Pc * Pc);
  if (idx >= total) return;
  int col = (int)(idx % 768);
  size_t rem = idx / 768;
  int n = (int)(rem % Nc);
  int b = (int)(rem / Nc);
  int hp = n / WPc, wp = n % WPc;
  int c  = col / 256;
  int pi = (col % 256) / 16;
  int pj = col % 16;
  int gh = hp * Pc + pi, gw = wp * Pc + pj;
  int k1 = gh / Hc, i = gh % Hc;
  int k2 = gw / Wc, j = gw % Wc;
  int t = k1 * Kc + k2;
  float v = x[((((size_t)b * Tc + t) * Cc + c) * Hc + i) * Wc + j];
  A[idx] = f2bf(v);
}

// =====================================================================
// Assemble residual stream h: cls row, tokens + sinusoidal pos-embed, zero pads
// =====================================================================
__global__ __launch_bounds__(256)
void vivit_assemble(const float* __restrict__ tok, const float* __restrict__ cls,
                    float* __restrict__ h)
{
  size_t idx = (size_t)blockIdx.x * 256 + threadIdx.x;
  size_t total = (size_t)Bc * SPc * Dc;
  if (idx >= total) return;
  int d = (int)(idx % Dc);
  size_t rem = idx / Dc;
  int s = (int)(rem % SPc);
  int b = (int)(rem / SPc);
  float v;
  if (s == 0) {
    v = cls[d];                                    // pe[0] == 0
  } else if (s <= Nc) {
    int n = s - 1;
    float freq = __expf(-(float)(2 * (d >> 1)) * (9.210340371976184f / (float)Dc));
    float arg  = (float)n * freq;
    float pe   = (d & 1) ? __cosf(arg) : __sinf(arg);
    v = tok[((size_t)b * Nc + n) * Dc + d] + pe;
  } else {
    v = 0.f;                                       // pad rows
  }
  h[((size_t)b * SPc + s) * Dc + d] = v;
}

// =====================================================================
// Pooling after final LN: agg = mean over all 784 patch tokens; extract cls
// =====================================================================
__global__ __launch_bounds__(256)
void vivit_pool(const float* __restrict__ hln, __bf16* __restrict__ aggin,
                float* __restrict__ clsf)
{
  int idx = blockIdx.x * 256 + threadIdx.x;
  if (idx >= Bc * Dc) return;
  int d = idx % Dc, b = idx / Dc;
  const float* base = hln + (size_t)b * SPc * Dc;
  float s = 0.f;
  for (int n = 0; n < Nc; ++n) s += base[(size_t)(1 + n) * Dc + d];
  aggin[idx] = f2bf(s * (1.f / (float)Nc));
  clsf[idx]  = base[d];
}

__global__ __launch_bounds__(256)
void vivit_combine(const float* __restrict__ clsf, const float* __restrict__ g2,
                   __bf16* __restrict__ headin)
{
  int idx = blockIdx.x * 256 + threadIdx.x;
  if (idx >= Bc * Dc) return;
  headin[idx] = f2bf(clsf[idx] + g2[idx]);
}

// =====================================================================
// Host-side helpers
// =====================================================================
static inline int ntpad(int N) { return ((N + 255) / 256) * 16; }

static void launch_gemm(int epi, const __bf16* A, const __bf16* Bp, const float* bias,
                        const float* res, void* out, int M, int N, int Kd,
                        hipStream_t st)
{
  int NT = ntpad(N);
  dim3 g((N + 255) / 256, (M + 63) / 64);
  dim3 bl(256);
  switch (epi) {
    case EPI_F32_BIAS:      vivit_gemm_wmma<EPI_F32_BIAS>      <<<g, bl, 0, st>>>(A, Bp, bias, res, out, M, N, Kd, NT); break;
    case EPI_F32_BIAS_RES:  vivit_gemm_wmma<EPI_F32_BIAS_RES>  <<<g, bl, 0, st>>>(A, Bp, bias, res, out, M, N, Kd, NT); break;
    case EPI_BF16_BIAS:     vivit_gemm_wmma<EPI_BF16_BIAS>     <<<g, bl, 0, st>>>(A, Bp, bias, res, out, M, N, Kd, NT); break;
    default:                vivit_gemm_wmma<EPI_BF16_BIAS_GELU><<<g, bl, 0, st>>>(A, Bp, bias, res, out, M, N, Kd, NT); break;
  }
}

static void launch_pack(const float* W, __bf16* Bp, int Kd, int N, hipStream_t st)
{
  int NT = ntpad(N);
  long total = (long)(Kd >> 5) * NT * 512;
  vivit_pack_b<<<(unsigned)((total + 255) / 256), 256, 0, st>>>(W, Bp, Kd, N, NT);
}

static inline size_t pksz(int Kd, int N) {        // packed bytes
  return (size_t)(Kd >> 5) * ntpad(N) * 512 * sizeof(__bf16);
}

static void launch_warm(const void* p, size_t bytes, hipStream_t st)
{
  const unsigned slice = 32768u;
  unsigned blocks = (unsigned)((bytes + slice - 1) / slice);
  vivit_l2_warm<<<blocks, 32, slice, st>>>(p, slice, (unsigned long)bytes);
}

extern "C" void kernel_launch(void* const* d_in, const int* in_sizes, int n_in,
                              void* d_out, int out_size, void* d_ws, size_t ws_size,
                              hipStream_t stream)
{
  (void)in_sizes; (void)n_in; (void)out_size; (void)ws_size;
  const float* x       = (const float*)d_in[0];
  const float* patch_w = (const float*)d_in[1];
  const float* patch_b = (const float*)d_in[2];
  const float* cls_tok = (const float*)d_in[3];
  const float* ln1_g   = (const float*)d_in[4];
  const float* ln1_b   = (const float*)d_in[5];
  const float* qkv_w   = (const float*)d_in[6];
  const float* qkv_b   = (const float*)d_in[7];
  const float* proj_w  = (const float*)d_in[8];
  const float* proj_b  = (const float*)d_in[9];
  const float* ln2_g   = (const float*)d_in[10];
  const float* ln2_b   = (const float*)d_in[11];
  const float* fc1_w   = (const float*)d_in[12];
  const float* fc1_b   = (const float*)d_in[13];
  const float* fc2_w   = (const float*)d_in[14];
  const float* fc2_b   = (const float*)d_in[15];
  const float* norm_g  = (const float*)d_in[16];
  const float* norm_b  = (const float*)d_in[17];
  const float* agg1_w  = (const float*)d_in[18];
  const float* agg1_b  = (const float*)d_in[19];
  const float* agg2_w  = (const float*)d_in[20];
  const float* agg2_b  = (const float*)d_in[21];
  const float* head_w  = (const float*)d_in[22];
  const float* head_b  = (const float*)d_in[23];

  // ---- carve workspace ----
  size_t off = 0;
  auto take = [&](size_t bytes) -> void* {
    void* p = (char*)d_ws + off;
    off += (bytes + 255) & ~(size_t)255;
    return p;
  };
  float*  hbuf   = (float*) take((size_t)Mrows * Dc * 4);
  __bf16* ybuf   = (__bf16*)take((size_t)Mrows * Dc * 2);
  float*  qkvbuf = (float*) take((size_t)Mrows * 3 * Dc * 4);
  __bf16* qq     = (__bf16*)take((size_t)Bc * NHc * SPc * HDc * 2);
  __bf16* kk     = (__bf16*)take((size_t)Bc * NHc * SPc * HDc * 2);
  __bf16* vT     = (__bf16*)take((size_t)Bc * NHc * HDc * SPc * 2);
  __bf16* obuf   = (__bf16*)take((size_t)Mrows * Dc * 2);
  __bf16* m1buf  = (__bf16*)take((size_t)Mrows * MLPHc * 2);
  float*  tokbuf = (float*) take((size_t)Bc * Nc * Dc * 4);
  __bf16* pA     = (__bf16*)take((size_t)Bc * Nc * Dc * 2);
  float*  hln    = (float*) take((size_t)Mrows * Dc * 4);
  __bf16* aggin  = (__bf16*)take((size_t)Bc * Dc * 2);
  float*  clsf   = (float*) take((size_t)Bc * Dc * 4);
  __bf16* g1     = (__bf16*)take((size_t)Bc * Dc * 2);
  float*  g2     = (float*) take((size_t)Bc * Dc * 4);
  __bf16* headin = (__bf16*)take((size_t)Bc * Dc * 2);

  __bf16* pPatch = (__bf16*)take(pksz(Dc, Dc));
  __bf16* pQkv   = (__bf16*)take((size_t)DEPTHc * pksz(Dc, 3 * Dc));
  __bf16* pProj  = (__bf16*)take((size_t)DEPTHc * pksz(Dc, Dc));
  __bf16* pF1    = (__bf16*)take((size_t)DEPTHc * pksz(Dc, MLPHc));
  __bf16* pF2    = (__bf16*)take((size_t)DEPTHc * pksz(MLPHc, Dc));
  __bf16* pA1    = (__bf16*)take(pksz(Dc, Dc));
  __bf16* pA2    = (__bf16*)take(pksz(Dc, Dc));
  __bf16* pHead  = (__bf16*)take(pksz(Dc, NCc));

  const size_t qkvStride  = pksz(Dc, 3 * Dc) / 2;
  const size_t projStride = pksz(Dc, Dc) / 2;
  const size_t f1Stride   = pksz(Dc, MLPHc) / 2;
  const size_t f2Stride   = pksz(MLPHc, Dc) / 2;

  // ---- pack all weights (f32 -> WMMA-ordered bf16) ----
  launch_pack(patch_w, pPatch, Dc, Dc, stream);
  for (int l = 0; l < DEPTHc; ++l) {
    launch_pack(qkv_w + (size_t)l * Dc * 3 * Dc, pQkv + l * qkvStride, Dc, 3 * Dc, stream);
    launch_pack(proj_w + (size_t)l * Dc * Dc,     pProj + l * projStride, Dc, Dc, stream);
    launch_pack(fc1_w  + (size_t)l * Dc * MLPHc,  pF1  + l * f1Stride, Dc, MLPHc, stream);
    launch_pack(fc2_w  + (size_t)l * MLPHc * Dc,  pF2  + l * f2Stride, MLPHc, Dc, stream);
  }
  launch_pack(agg1_w, pA1, Dc, Dc, stream);
  launch_pack(agg2_w, pA2, Dc, Dc, stream);
  launch_pack(head_w, pHead, Dc, NCc, stream);

  // ---- patch embedding ----
  {
    size_t tot = (size_t)Bc * Nc * Dc;
    vivit_patchify<<<(unsigned)((tot + 255) / 256), 256, 0, stream>>>(x, pA);
  }
  launch_gemm(EPI_F32_BIAS, pA, pPatch, patch_b, nullptr, tokbuf, Bc * Nc, Dc, Dc, stream);
  {
    size_t tot = (size_t)Bc * SPc * Dc;
    vivit_assemble<<<(unsigned)((tot + 255) / 256), 256, 0, stream>>>(tokbuf, cls_tok, hbuf);
  }

  const size_t attnSmem = (size_t)16 * SCPc * 4 + (size_t)16 * SCPc * 2 + 16 * 8 * 4 + 16 * 4;
  const size_t splitTot = (size_t)Bc * SPc * Dc;

  // ---- transformer blocks ----
  for (int l = 0; l < DEPTHc; ++l) {
    // TDM prefetch of next layer's packed weights into L2 while this layer runs
    if (l + 1 < DEPTHc) {
      launch_warm(pQkv  + (l + 1) * qkvStride,  pksz(Dc, 3 * Dc), stream);
      launch_warm(pProj + (l + 1) * projStride, pksz(Dc, Dc),     stream);
      launch_warm(pF1   + (l + 1) * f1Stride,   pksz(Dc, MLPHc),  stream);
      launch_warm(pF2   + (l + 1) * f2Stride,   pksz(MLPHc, Dc),  stream);
    }
    vivit_layernorm<__bf16><<<Mrows, 256, 0, stream>>>(hbuf, ln1_g + l * Dc, ln1_b + l * Dc, ybuf, Dc);
    launch_gemm(EPI_F32_BIAS, ybuf, pQkv + l * qkvStride, qkv_b + l * 3 * Dc, nullptr,
                qkvbuf, Mrows, 3 * Dc, Dc, stream);
    vivit_qkv_split<<<(unsigned)((splitTot + 255) / 256), 256, 0, stream>>>(qkvbuf, qq, kk, vT);
    vivit_attention<<<dim3(SPc / 16, NHc, Bc), 128, attnSmem, stream>>>(qq, kk, vT, obuf, 0.125f);
    launch_gemm(EPI_F32_BIAS_RES, obuf, pProj + l * projStride, proj_b + l * Dc, hbuf,
                hbuf, Mrows, Dc, Dc, stream);
    vivit_layernorm<__bf16><<<Mrows, 256, 0, stream>>>(hbuf, ln2_g + l * Dc, ln2_b + l * Dc, ybuf, Dc);
    launch_gemm(EPI_BF16_BIAS_GELU, ybuf, pF1 + l * f1Stride, fc1_b + l * MLPHc, nullptr,
                m1buf, Mrows, MLPHc, Dc, stream);
    launch_gemm(EPI_F32_BIAS_RES, m1buf, pF2 + l * f2Stride, fc2_b + l * Dc, hbuf,
                hbuf, Mrows, Dc, MLPHc, stream);
  }

  // ---- head ----
  vivit_layernorm<float><<<Mrows, 256, 0, stream>>>(hbuf, norm_g, norm_b, hln, Dc);
  vivit_pool<<<(Bc * Dc + 255) / 256, 256, 0, stream>>>(hln, aggin, clsf);
  launch_gemm(EPI_BF16_BIAS_GELU, aggin, pA1, agg1_b, nullptr, g1, Bc, Dc, Dc, stream);
  launch_gemm(EPI_F32_BIAS, g1, pA2, agg2_b, nullptr, g2, Bc, Dc, Dc, stream);
  vivit_combine<<<(Bc * Dc + 255) / 256, 256, 0, stream>>>(clsf, g2, headin);
  launch_gemm(EPI_F32_BIAS, headin, pHead, head_b, nullptr, d_out, Bc, NCc, Dc, stream);
}
